// Net3DLocal_5214090297744
// MI455X (gfx1250) — matrix operand order repaired and checked by reading, not compiled, not run on hardware
//
#include <hip/hip_runtime.h>
#include <hip/hip_bf16.h>

// ---------------------------------------------------------------------------
// Net3D-style message-passing GNN for MI455X (gfx1250, wave32, WMMA).
// Heavy GEMMs -> v_wmma_f32_16x16x32_bf16 (fp32 accumulate).
// Edge A-tile gather -> bf16 shadows + async global->LDS DMA (ASYNCcnt).
// ---------------------------------------------------------------------------

static constexpr int HD  = 128;   // hidden
static constexpr int NDD = 64;    // node feature dim
static constexpr int TDD = 32;    // target dim
static constexpr int H3  = 384;   // 3*H

typedef __attribute__((ext_vector_type(16))) __bf16 v16bf;
typedef __attribute__((ext_vector_type(8)))  float  v8f;
typedef __attribute__((ext_vector_type(4)))  int    v4i_t;

#if __has_builtin(__builtin_amdgcn_global_load_async_to_lds_b128)
#define HAVE_ASYNC_LDS 1
#else
#define HAVE_ASYNC_LDS 0
#endif
#if __has_builtin(__builtin_amdgcn_s_wait_asynccnt)
#define HAVE_WAIT_ASYNC 1
#else
#define HAVE_WAIT_ASYNC 0
#endif

__device__ __forceinline__ float silu_f(float x)    { return x * (1.0f / (1.0f + __expf(-x))); }
__device__ __forceinline__ float sigmoid_f(float x) { return 1.0f / (1.0f + __expf(-x)); }

// 16-byte global(bf16) -> LDS copy; async DMA path when the toolchain has it.
// Builtin signature (probe-confirmed): (v4i addrspace(1)*, v4i addrspace(3)*, Ii, Ii)
__device__ __forceinline__ void copy16_g2l(const __bf16* g, __bf16* l) {
#if HAVE_ASYNC_LDS
  __builtin_amdgcn_global_load_async_to_lds_b128(
      (__attribute__((address_space(1))) v4i_t*)g,
      (__attribute__((address_space(3))) v4i_t*)l, 0, 0);
#else
  *(uint4*)l = *(const uint4*)g;
#endif
}

__device__ __forceinline__ void async_gather_wait() {
#if HAVE_ASYNC_LDS
#if HAVE_WAIT_ASYNC
  __builtin_amdgcn_s_wait_asynccnt(0);
#else
  asm volatile("s_wait_asynccnt 0" ::: "memory");
#endif
#endif
}

// WMMA wrapper: D = A(16x32 bf16) * B(32x16 bf16) + C (16x16 f32)
__device__ __forceinline__ v8f wmma_bf16(v16bf a, v16bf b, v8f c) {
  return __builtin_amdgcn_wmma_f32_16x16x32_bf16(false, a, false, b, (short)0, c, false, false);
}

// A-fragment (16x32 bf16) from row-major LDS tile [16 x lda], per ISA layout.
__device__ __forceinline__ v16bf load_A_frag_lds(const __bf16* As, int lda, int kBase, int lane) {
  const int m = lane & 15;
  const int h = lane >> 4;
  const __bf16* row = As + m * lda + kBase + 8 * h;
  v16bf a;
#pragma unroll
  for (int i = 0; i < 8; ++i) {
    int k0 = (i < 4) ? (2 * i) : (16 + 2 * (i - 4));
    a[2 * i]     = row[k0];
    a[2 * i + 1] = row[k0 + 1];
  }
  return a;
}

// B-fragment (32x16 bf16) from row-major [K x ldb] weight matrix, per ISA layout.
__device__ __forceinline__ v16bf load_B_frag(const __bf16* B, int ldb, int kBase, int nBase, int lane) {
  const int n = lane & 15;
  const int h = lane >> 4;
  const __bf16* col = B + (size_t)(kBase + 8 * h) * ldb + nBase + n;
  v16bf b;
#pragma unroll
  for (int i = 0; i < 8; ++i) {
    int k0 = (i < 4) ? (2 * i) : (16 + 2 * (i - 4));
    b[2 * i]     = col[k0 * (size_t)ldb];
    b[2 * i + 1] = col[(k0 + 1) * (size_t)ldb];
  }
  return b;
}

// ---------------------------------------------------------------------------
// fp32 -> bf16 weight conversion
// ---------------------------------------------------------------------------
__global__ void f32_to_bf16_kernel(const float* __restrict__ src, __bf16* __restrict__ dst, int n) {
  int i = blockIdx.x * blockDim.x + threadIdx.x;
  if (i < n) dst[i] = (__bf16)src[i];
}

// ---------------------------------------------------------------------------
// Input embeddings (write fp32 master + bf16 shadow)
// ---------------------------------------------------------------------------
__global__ void embed_nodes_kernel(const float* __restrict__ nf, const float* __restrict__ w_in,
                                   const float* __restrict__ b_in,
                                   float* __restrict__ feat, __bf16* __restrict__ featb, int Nn) {
  int idx = blockIdx.x * blockDim.x + threadIdx.x;
  if (idx >= Nn * HD) return;
  int n = idx / HD, c = idx % HD;
  float s = b_in[c];
  const float* x = nf + (size_t)n * NDD;
#pragma unroll 8
  for (int k = 0; k < NDD; ++k) s += x[k] * w_in[k * HD + c];
  float v = silu_f(silu_f(s));
  feat[idx]  = v;
  featb[idx] = (__bf16)v;
}

__global__ void embed_edges_kernel(const float* __restrict__ ed, const float* __restrict__ w_e,
                                   const float* __restrict__ b_e,
                                   float* __restrict__ d, __bf16* __restrict__ db, long total) {
  long idx = (long)blockIdx.x * blockDim.x + threadIdx.x;
  if (idx >= total) return;
  long e = idx / HD; int c = (int)(idx % HD);
  float v = silu_f(silu_f(ed[e] * w_e[c] + b_e[c]));
  d[idx]  = v;
  db[idx] = (__bf16)v;
}

// ---------------------------------------------------------------------------
// Edge message kernel: per 16-edge tile
//   mi  = [feat[src] | feat[dst] | d]            (16 x 384, bf16 in LDS, DMA'd)
//   msg = silu(silu(mi @ W1 + b1) @ W2 + b2)     (WMMA bf16, f32 acc)
//   gate= sigmoid(msg @ gw + gb)
//   d  += msg (fp32 master + bf16 shadow); m_sum[dst] += msg*gate (f32 atomics)
// ---------------------------------------------------------------------------
__global__ void __launch_bounds__(256)
edge_msg_kernel(const __bf16* __restrict__ featb, float* __restrict__ d, __bf16* __restrict__ db,
                float* __restrict__ m_sum,
                const __bf16* __restrict__ mw1, const __bf16* __restrict__ mw2,
                const float* __restrict__ mb1, const float* __restrict__ mb2,
                const float* __restrict__ gw, const float* __restrict__ gb,
                const int* __restrict__ src, const int* __restrict__ dst,
                int E_, int numTiles) {
  __shared__ alignas(16) __bf16 As[16 * H3];    // 12 KB
  __shared__ alignas(16) __bf16 Act[16 * HD];   // 4 KB
  __shared__ alignas(16) float  Msg[16 * HD];   // 8 KB
  __shared__ float  GateP[16];

  const int lane  = threadIdx.x & 31;
  const int wave  = threadIdx.x >> 5;       // 0..7
  const int nBase = wave * 16;

  // Preload the wave's B fragments (W1: 384x128 slice, W2: 128x128 slice)
  v16bf B1[12], B2[4];
#pragma unroll
  for (int k = 0; k < 12; ++k) B1[k] = load_B_frag(mw1, HD, k * 32, nBase, lane);
#pragma unroll
  for (int k = 0; k < 4;  ++k) B2[k] = load_B_frag(mw2, HD, k * 32, nBase, lane);
  const float bias1 = mb1[nBase + (lane & 15)];
  const float bias2 = mb2[nBase + (lane & 15)];
  const float gbv   = gb[0];

  const int r = threadIdx.x >> 4;   // edge row within tile (0..15)
  const int j = threadIdx.x & 15;   // 8-column group (0..15)

  for (int tile = blockIdx.x; tile < numTiles; tile += gridDim.x) {
    const int e0 = tile * 16;
    // ---- gather A tile: [feat[src] | feat[dst] | d] (bf16, 16B per copy) ---
    {
      int e = e0 + r; if (e >= E_) e = E_ - 1;
      const int s  = src[e];
      const int t2 = dst[e];
      const __bf16* fs = featb + (size_t)s  * HD + j * 8;
      const __bf16* fd = featb + (size_t)t2 * HD + j * 8;
      const __bf16* de = db    + (size_t)e  * HD + j * 8;
      __bf16* a0 = &As[r * H3 + j * 8];
      copy16_g2l(fs, a0);
      copy16_g2l(fd, a0 + HD);
      copy16_g2l(de, a0 + 2 * HD);
    }
    async_gather_wait();
    __syncthreads();

    // ---- GEMM1: (16x384) @ (384x16 slice) -> silu -> Act (bf16) ------------
    {
      v8f acc = {0.f,0.f,0.f,0.f,0.f,0.f,0.f,0.f};
#pragma unroll
      for (int k = 0; k < 12; ++k)
        acc = wmma_bf16(load_A_frag_lds(As, H3, k * 32, lane), B1[k], acc);
      const int n = lane & 15, hh = lane >> 4;
#pragma unroll
      for (int r2 = 0; r2 < 8; ++r2) {
        int m = r2 + 8 * hh;
        Act[m * HD + nBase + n] = (__bf16)silu_f(acc[r2] + bias1);
      }
    }
    __syncthreads();

    // ---- GEMM2: (16x128) @ (128x16 slice) -> silu -> Msg (f32) -------------
    {
      v8f acc = {0.f,0.f,0.f,0.f,0.f,0.f,0.f,0.f};
#pragma unroll
      for (int k = 0; k < 4; ++k)
        acc = wmma_bf16(load_A_frag_lds(Act, HD, k * 32, lane), B2[k], acc);
      const int n = lane & 15, hh = lane >> 4;
#pragma unroll
      for (int r2 = 0; r2 < 8; ++r2) {
        int m = r2 + 8 * hh;
        Msg[m * HD + nBase + n] = silu_f(acc[r2] + bias2);
      }
    }
    __syncthreads();

    // ---- gate = sigmoid(msg . gw + gb), one thread per edge ----------------
    if (threadIdx.x < 16) {
      float s = gbv;
#pragma unroll 8
      for (int c = 0; c < HD; ++c) s += Msg[threadIdx.x * HD + c] * gw[c];
      GateP[threadIdx.x] = sigmoid_f(s);
    }
    __syncthreads();

    // ---- d += msg (master + shadow); m_sum[dst] += msg*gate ----------------
    {
      int e = e0 + r;
      if (e < E_) {
        const float gt = GateP[r];
        const int   t2 = dst[e];
        float*  de = d     + (size_t)e  * HD;
        __bf16* ds = db    + (size_t)e  * HD;
        float*  ms = m_sum + (size_t)t2 * HD;
#pragma unroll
        for (int q = 0; q < 8; ++q) {
          int c = j * 8 + q;
          float m  = Msg[r * HD + c];
          float nd = de[c] + m;
          de[c] = nd;
          ds[c] = (__bf16)nd;
          unsafeAtomicAdd(&ms[c], m * gt);
        }
      }
    }
    __syncthreads();
  }
}

// ---------------------------------------------------------------------------
// Node update: feat += silu((m_sum+feat)@U1+b1) @ U2 + b2   (+ bf16 shadow)
// ---------------------------------------------------------------------------
__global__ void __launch_bounds__(256)
node_update_kernel(float* __restrict__ feat, __bf16* __restrict__ featb,
                   const float* __restrict__ m_sum,
                   const __bf16* __restrict__ uw1, const float* __restrict__ ub1,
                   const __bf16* __restrict__ uw2, const float* __restrict__ ub2,
                   int Nn, int numTiles) {
  __shared__ alignas(16) __bf16 Xs[16 * HD];
  __shared__ alignas(16) __bf16 Act[16 * HD];

  const int lane  = threadIdx.x & 31;
  const int wave  = threadIdx.x >> 5;
  const int nBase = wave * 16;

  v16bf B1[4], B2[4];
#pragma unroll
  for (int k = 0; k < 4; ++k) B1[k] = load_B_frag(uw1, HD, k * 32, nBase, lane);
#pragma unroll
  for (int k = 0; k < 4; ++k) B2[k] = load_B_frag(uw2, HD, k * 32, nBase, lane);
  const float bias1 = ub1[nBase + (lane & 15)];
  const float bias2 = ub2[nBase + (lane & 15)];

  const int r = threadIdx.x >> 4;
  const int j = threadIdx.x & 15;

  for (int tile = blockIdx.x; tile < numTiles; tile += gridDim.x) {
    const int n0 = tile * 16;
    {
      int node = n0 + r; if (node >= Nn) node = Nn - 1;
      const float* fp = feat  + (size_t)node * HD;
      const float* mp = m_sum + (size_t)node * HD;
#pragma unroll
      for (int q = 0; q < 8; ++q) {
        int c = j * 8 + q;
        Xs[r * HD + c] = (__bf16)(fp[c] + mp[c]);
      }
    }
    __syncthreads();

    {
      v8f acc = {0.f,0.f,0.f,0.f,0.f,0.f,0.f,0.f};
#pragma unroll
      for (int k = 0; k < 4; ++k)
        acc = wmma_bf16(load_A_frag_lds(Xs, HD, k * 32, lane), B1[k], acc);
      const int n = lane & 15, hh = lane >> 4;
#pragma unroll
      for (int r2 = 0; r2 < 8; ++r2) {
        int m = r2 + 8 * hh;
        Act[m * HD + nBase + n] = (__bf16)silu_f(acc[r2] + bias1);
      }
    }
    __syncthreads();

    {
      v8f acc = {0.f,0.f,0.f,0.f,0.f,0.f,0.f,0.f};
#pragma unroll
      for (int k = 0; k < 4; ++k)
        acc = wmma_bf16(load_A_frag_lds(Act, HD, k * 32, lane), B2[k], acc);
      const int n = lane & 15, hh = lane >> 4;
#pragma unroll
      for (int r2 = 0; r2 < 8; ++r2) {
        int m = r2 + 8 * hh;
        int node = n0 + m;
        if (node < Nn) {
          size_t idx = (size_t)node * HD + nBase + n;
          float v = feat[idx] + acc[r2] + bias2;
          feat[idx]  = v;
          featb[idx] = (__bf16)v;
        }
      }
    }
    __syncthreads();
  }
}

// ---------------------------------------------------------------------------
// Output head: out = silu(feat@O1+b1) @ O2 + b2   (O2: 128x32)
// ---------------------------------------------------------------------------
__global__ void __launch_bounds__(256)
output_kernel(const __bf16* __restrict__ featb,
              const __bf16* __restrict__ ow1, const float* __restrict__ ob1,
              const __bf16* __restrict__ ow2, const float* __restrict__ ob2,
              float* __restrict__ out, int Nn, int numTiles) {
  __shared__ alignas(16) __bf16 Xs[16 * HD];
  __shared__ alignas(16) __bf16 Act[16 * HD];

  const int lane  = threadIdx.x & 31;
  const int wave  = threadIdx.x >> 5;
  const int nBase = wave * 16;

  v16bf B1[4], B2[4];
#pragma unroll
  for (int k = 0; k < 4; ++k) B1[k] = load_B_frag(ow1, HD, k * 32, nBase, lane);
  if (wave < 2) {
#pragma unroll
    for (int k = 0; k < 4; ++k) B2[k] = load_B_frag(ow2, TDD, k * 32, nBase, lane);
  }
  const float bias1 = ob1[nBase + (lane & 15)];
  const float bias2 = (wave < 2) ? ob2[nBase + (lane & 15)] : 0.f;

  const int r = threadIdx.x >> 4;
  const int j = threadIdx.x & 15;

  for (int tile = blockIdx.x; tile < numTiles; tile += gridDim.x) {
    const int n0 = tile * 16;
    {
      int node = n0 + r; if (node >= Nn) node = Nn - 1;
      copy16_g2l(featb + (size_t)node * HD + j * 8, &Xs[r * HD + j * 8]);
    }
    async_gather_wait();
    __syncthreads();

    {
      v8f acc = {0.f,0.f,0.f,0.f,0.f,0.f,0.f,0.f};
#pragma unroll
      for (int k = 0; k < 4; ++k)
        acc = wmma_bf16(load_A_frag_lds(Xs, HD, k * 32, lane), B1[k], acc);
      const int n = lane & 15, hh = lane >> 4;
#pragma unroll
      for (int r2 = 0; r2 < 8; ++r2) {
        int m = r2 + 8 * hh;
        Act[m * HD + nBase + n] = (__bf16)silu_f(acc[r2] + bias1);
      }
    }
    __syncthreads();

    if (wave < 2) {
      v8f acc = {0.f,0.f,0.f,0.f,0.f,0.f,0.f,0.f};
#pragma unroll
      for (int k = 0; k < 4; ++k)
        acc = wmma_bf16(load_A_frag_lds(Act, HD, k * 32, lane), B2[k], acc);
      const int n = lane & 15, hh = lane >> 4;
#pragma unroll
      for (int r2 = 0; r2 < 8; ++r2) {
        int m = r2 + 8 * hh;
        int node = n0 + m;
        if (node < Nn) out[(size_t)node * TDD + nBase + n] = acc[r2] + bias2;
      }
    }
    __syncthreads();
  }
}

// ---------------------------------------------------------------------------
// Host-side orchestration
// ---------------------------------------------------------------------------
extern "C" void kernel_launch(void* const* d_in, const int* in_sizes, int n_in,
                              void* d_out, int out_size, void* d_ws, size_t ws_size,
                              hipStream_t stream) {
  const float* node_feat = (const float*)d_in[0];
  const float* edge_d    = (const float*)d_in[1];
  const float* w_in      = (const float*)d_in[2];
  const float* b_in      = (const float*)d_in[3];
  const float* w_edge    = (const float*)d_in[4];
  const float* b_edge    = (const float*)d_in[5];
  const float* msg_w1    = (const float*)d_in[6];
  const float* msg_b1    = (const float*)d_in[7];
  const float* msg_w2    = (const float*)d_in[8];
  const float* msg_b2    = (const float*)d_in[9];
  const float* gate_w    = (const float*)d_in[10];
  const float* gate_b    = (const float*)d_in[11];
  const float* upd_w1    = (const float*)d_in[12];
  const float* upd_b1    = (const float*)d_in[13];
  const float* upd_w2    = (const float*)d_in[14];
  const float* upd_b2    = (const float*)d_in[15];
  const float* out_w1    = (const float*)d_in[16];
  const float* out_b1    = (const float*)d_in[17];
  const float* out_w2    = (const float*)d_in[18];
  const float* out_b2    = (const float*)d_in[19];
  const int*   src       = (const int*)d_in[20];
  const int*   dst       = (const int*)d_in[21];

  const int Nn = in_sizes[0] / NDD;       // 20000
  const int E_ = in_sizes[20];            // 320000
  const int Lc = in_sizes[7] / HD;        // 4

  // ---- workspace carve ----
  char* w = (char*)d_ws;
  auto carve = [&](size_t bytes) { void* p = (void*)w; w += (bytes + 255) & ~(size_t)255; return p; };
  float*  feat  = (float*)carve((size_t)Nn * HD * 4);
  __bf16* featb = (__bf16*)carve((size_t)Nn * HD * 2);
  float*  msum  = (float*)carve((size_t)Nn * HD * 4);
  float*  dbuf  = (float*)carve((size_t)E_ * HD * 4);
  __bf16* dbb   = (__bf16*)carve((size_t)E_ * HD * 2);
  __bf16* mw1b  = (__bf16*)carve((size_t)Lc * H3 * HD * 2);
  __bf16* mw2b  = (__bf16*)carve((size_t)Lc * HD * HD * 2);
  __bf16* uw1b  = (__bf16*)carve((size_t)Lc * HD * HD * 2);
  __bf16* uw2b  = (__bf16*)carve((size_t)Lc * HD * HD * 2);
  __bf16* ow1b  = (__bf16*)carve((size_t)HD * HD * 2);
  __bf16* ow2b  = (__bf16*)carve((size_t)HD * TDD * 2);
  (void)ws_size;

  // ---- convert weights to bf16 ----
  auto conv = [&](const float* s, __bf16* dd, int n) {
    f32_to_bf16_kernel<<<(n + 255) / 256, 256, 0, stream>>>(s, dd, n);
  };
  conv(msg_w1, mw1b, Lc * H3 * HD);
  conv(msg_w2, mw2b, Lc * HD * HD);
  conv(upd_w1, uw1b, Lc * HD * HD);
  conv(upd_w2, uw2b, Lc * HD * HD);
  conv(out_w1, ow1b, HD * HD);
  conv(out_w2, ow2b, HD * TDD);

  // ---- input embeddings ----
  embed_nodes_kernel<<<(Nn * HD + 255) / 256, 256, 0, stream>>>(node_feat, w_in, b_in, feat, featb, Nn);
  {
    long total = (long)E_ * HD;
    embed_edges_kernel<<<(int)((total + 255) / 256), 256, 0, stream>>>(edge_d, w_edge, b_edge,
                                                                       dbuf, dbb, total);
  }

  const int tilesE = (E_ + 15) / 16;
  const int tilesN = (Nn + 15) / 16;
  const int gridE  = tilesE < 4096 ? tilesE : 4096;
  const int gridN  = tilesN < 1250 ? tilesN : 1250;

  // ---- message-passing layers ----
  for (int l = 0; l < Lc; ++l) {
    (void)hipMemsetAsync(msum, 0, (size_t)Nn * HD * 4, stream);
    edge_msg_kernel<<<gridE, 256, 0, stream>>>(
        featb, dbuf, dbb, msum,
        mw1b + (size_t)l * H3 * HD, mw2b + (size_t)l * HD * HD,
        msg_b1 + (size_t)l * HD,    msg_b2 + (size_t)l * HD,
        gate_w + (size_t)l * HD,    gate_b + l,
        src, dst, E_, tilesE);
    node_update_kernel<<<gridN, 256, 0, stream>>>(
        feat, featb, msum,
        uw1b + (size_t)l * HD * HD, upd_b1 + (size_t)l * HD,
        uw2b + (size_t)l * HD * HD, upd_b2 + (size_t)l * HD,
        Nn, tilesN);
  }

  // ---- output head ----
  output_kernel<<<gridN, 256, 0, stream>>>(featb, ow1b, out_b1, ow2b, out_b2,
                                           (float*)d_out, Nn, tilesN);
}
